// BodyModel_53841710022832
// MI455X (gfx1250) — matrix operand order repaired and checked by reading, not compiled, not run on hardware
//
#include <hip/hip_runtime.h>
#include <hip/hip_bf16.h>
#include <math.h>

#define BSZ 32768

typedef __attribute__((ext_vector_type(16))) __bf16 v16bf;
typedef __attribute__((ext_vector_type(8)))  float  v8f;

union BF16Frag { v16bf v; __bf16 h[16]; unsigned short u[16]; };

// software RNE fallback (used only by weight prep, not in the GEMM hot loop)
__device__ __forceinline__ unsigned short f2bf(float f) {
  union { float f; unsigned int u; } x; x.f = f;
  unsigned int u = x.u;
  u += 0x7fffu + ((u >> 16) & 1u);
  return (unsigned short)(u >> 16);
}

__device__ __forceinline__ float sigmoidf_(float x) {
  return 1.0f / (1.0f + __expf(-x));
}

// ---------------------------------------------------------------------------
// Weight prep: fp32 [K,N] (optionally transposed source) -> bf16 swizzled into
// 32x16 WMMA B fragments in contiguous K order (as A): lane<16 holds col n=lane&15, K e=0..15 of the
// k-tile; lane>=16 holds the same col, K 16..31. 16 elements/lane contiguous.
// dst index = (((n_tile * kTiles) + k_tile) * 32 + lane) * 16 + e
// ---------------------------------------------------------------------------
__global__ void prep_weight(const float* __restrict__ src,
                            unsigned short* __restrict__ dst,
                            int Keff, int Kpad, int Npad, int Neff,
                            int srcStride, int transpose) {
  int t = blockIdx.x * blockDim.x + threadIdx.x;
  int kTiles = Kpad >> 5;
  int total = kTiles * (Npad >> 4) * 512;
  if (t >= total) return;
  int e    = t & 15;
  int lane = (t >> 4) & 31;
  int rest = t >> 9;
  int kt = rest % kTiles;
  int nt = rest / kTiles;
  int n = (nt << 4) + (lane & 15);
  int k = (kt << 5) + ((lane & 16) ? 16 : 0) + e;
  float v = 0.0f;
  if (k < Keff && n < Neff)
    v = transpose ? src[(size_t)n * srcStride + k]
                  : src[(size_t)k * srcStride + n];
  dst[t] = f2bf(v);
}

// ---------------------------------------------------------------------------
// Generic GEMM: C[M,Nout] = act(A[M,Kpad] * Bsw + bias).
// NT = compile-time n-tiles per wave (wave owns a 16 x NT*16 C strip); the
// single A fragment is reused across NT fully-unrolled WMMAs per K-step, so
// accumulators stay in statically-indexed VGPRs (no v_movrel). A conversion
// uses native __bf16 casts -> v_cvt*bf16_f32 hardware converts.
// ---------------------------------------------------------------------------
template <int NT>
__global__ __launch_bounds__(256) void gemm_bf16_wmma(
    const float* __restrict__ A, const unsigned short* __restrict__ Bsw,
    const float* __restrict__ bias, float* __restrict__ C,
    int M, int Kpad, int Npad, int Nout, int lda, int ldc, int act) {
  const int lane = threadIdx.x & 31;
  const int mTiles  = M >> 4;
  const int nGroups = Npad / (NT * 16);       // Npad is a multiple of NT*16
  const long gw = (long)blockIdx.x * (blockDim.x >> 5) + (threadIdx.x >> 5);
  if (gw >= (long)mTiles * nGroups) return;   // wave-uniform
  const int mTile = (int)(gw / nGroups);
  const int ng    = (int)(gw % nGroups);

  const int row  = (mTile << 4) + (lane & 15);
  const int kOff = (lane & 16) ? 16 : 0;      // contiguous K layout (matches Bsw)
  const float* aRow = A + (size_t)row * lda;

  v8f z = {};
  v8f acc[NT];
#pragma unroll
  for (int i = 0; i < NT; ++i) acc[i] = z;
  const int kTiles = Kpad >> 5;

  for (int kt = 0; kt < kTiles; ++kt) {
    const float* ap = aRow + (kt << 5) + kOff;
    __builtin_prefetch(ap + 32, 0, 0);        // global_prefetch_b8 (next K tile)
    BF16Frag af;
#pragma unroll
    for (int e = 0; e < 8; ++e) af.h[e]     = (__bf16)ap[e];      // K 0..7 / 16..23
#pragma unroll
    for (int e = 0; e < 8; ++e) af.h[8 + e] = (__bf16)ap[8 + e];  // K 8..15 / 24..31
#pragma unroll
    for (int nt = 0; nt < NT; ++nt) {
      const int n0 = ng * NT + nt;
      const unsigned short* bp =
          Bsw + ((((size_t)n0 * kTiles) + kt) * 32 + lane) * 16;
      v16bf bf = *(const v16bf*)bp;           // one 32B fragment load
      acc[nt] = __builtin_amdgcn_wmma_f32_16x16x32_bf16(
          false, af.v, false, bf, (short)0, acc[nt], false, false);
    }
  }

  const int rbase = (mTile << 4) + ((lane & 16) ? 8 : 0);
#pragma unroll
  for (int nt = 0; nt < NT; ++nt) {
    const int col = ((ng * NT + nt) << 4) + (lane & 15);
    if (col >= Nout) continue;                // per-lane, post-WMMA only
    const float bb = bias ? bias[col] : 0.0f;
#pragma unroll
    for (int i = 0; i < 8; ++i) {
      float v = acc[nt][i] + bb;
      if (act == 1) v = fmaxf(v, 0.0f);
      C[(size_t)(rbase + i) * ldc + col] = v;
    }
  }
}

// ---------------------------------------------------------------------------
// Encoder: per (b, pos) produce 32 contiguous features
// [car4 queue4 occ4 flow4 stop4 press4 phase4 mask4] -> ENC[b, pos*32 + j]
// ---------------------------------------------------------------------------
struct EncArgs {
  const float* x[6];
  const float* w[6];
  const float* b[6];
  const float* phase_emb;
  const float* mask_emb;
  const int*   cur;
  const int*   msk;
  float*       enc;
};

__global__ void encode_kernel(EncArgs a) {
  int t = blockIdx.x * blockDim.x + threadIdx.x;
  if (t >= BSZ * 224) return;
  int j  = t & 31;
  int bp = t >> 5;          // b*7 + pos
  int g  = j >> 2;
  int jj = j & 3;
  float v;
  if (g < 6) {
    v = sigmoidf_(a.x[g][bp] * a.w[g][jj] + a.b[g][jj]);
  } else if (g == 6) {
    v = sigmoidf_(a.phase_emb[a.cur[bp] * 4 + jj]);
  } else {
    v = sigmoidf_(a.mask_emb[a.msk[bp] * 4 + jj]);
  }
  a.enc[t] = v;
}

// ---------------------------------------------------------------------------
// GRU gate math (torch GRUCell semantics)
// ---------------------------------------------------------------------------
__global__ void gru_kernel(const float* __restrict__ gi,
                           const float* __restrict__ gh,
                           const float* __restrict__ hprev,
                           float* __restrict__ h) {
  int t = blockIdx.x * blockDim.x + threadIdx.x;
  if (t >= BSZ * 256) return;
  int b = t >> 8, c = t & 255;
  size_t base = (size_t)b * 768;
  float ir = gi[base + c], iz = gi[base + 256 + c], ig = gi[base + 512 + c];
  float hr = gh[base + c], hz = gh[base + 256 + c], hg = gh[base + 512 + c];
  float r  = sigmoidf_(ir + hr);
  float zz = sigmoidf_(iz + hz);
  float n  = tanhf(ig + r * hg);
  h[t] = (1.0f - zz) * n + zz * hprev[t];
}

// ---------------------------------------------------------------------------
// Neighbor gather: NBCAT[b*4+n, 288] = [h[idx] (256) | sigmoid(dis*wd+bd) (4)
//                                      | eye4 row n (4) | zeros (24 pad)]
// ---------------------------------------------------------------------------
__global__ void gather_kernel(const float* __restrict__ h,
                              const int* __restrict__ nbi,
                              const float* __restrict__ ndis,
                              const float* __restrict__ w_dis,
                              const float* __restrict__ b_dis,
                              float* __restrict__ nbcat) {
  int t = blockIdx.x * blockDim.x + threadIdx.x;
  if (t >= BSZ * 4 * 288) return;
  int c  = t % 288;
  int r  = t / 288;
  int b  = r >> 2;
  int nn = r & 3;
  float v = 0.0f;
  if (c < 256) {
    int idx = nbi[b * 4 + nn];
    v = (idx < 0) ? 0.0f : h[(size_t)idx * 256 + c];
  } else if (c < 260) {
    int j = c - 256;
    v = sigmoidf_(ndis[b * 4 + nn] * w_dis[j] + b_dis[j]);
  } else if (c < 264) {
    v = ((c - 260) == nn) ? 1.0f : 0.0f;
  }
  nbcat[t] = v;
}

// ---------------------------------------------------------------------------
// Attention weights: one thread per (b, head) -> 4 masked scores + softmax
// ---------------------------------------------------------------------------
__global__ void attn_kernel(const float* __restrict__ Q,
                            const float* __restrict__ K,
                            const int* __restrict__ nbi,
                            float* __restrict__ attnw) {
  int t = blockIdx.x * blockDim.x + threadIdx.x;
  if (t >= BSZ * 4) return;
  int b = t >> 2, hh = t & 3;
  const float* qp = Q + (size_t)b * 512 + hh * 128;
  float s[4];
#pragma unroll
  for (int n = 0; n < 4; ++n) {
    int idx = nbi[b * 4 + n];
    if (idx < 0) { s[n] = -1e9f; continue; }
    const float* kp = K + ((size_t)b * 4 + n) * 512 + hh * 128;
    float acc = 0.0f;
    for (int d = 0; d < 128; ++d) acc = fmaf(qp[d], kp[d], acc);
    s[n] = acc * 0.08838834764831845f;   // 1/sqrt(128)
  }
  float m = fmaxf(fmaxf(s[0], s[1]), fmaxf(s[2], s[3]));
  float e0 = __expf(s[0] - m), e1 = __expf(s[1] - m);
  float e2 = __expf(s[2] - m), e3 = __expf(s[3] - m);
  float inv = 1.0f / (e0 + e1 + e2 + e3);
  attnw[t * 4 + 0] = e0 * inv;
  attnw[t * 4 + 1] = e1 * inv;
  attnw[t * 4 + 2] = e2 * inv;
  attnw[t * 4 + 3] = e3 * inv;
}

// ---------------------------------------------------------------------------
// Context: ctx[b, hh*128+d] = sum_n attn[b,hh,n] * V[b*4+n, hh*128+d]
// ---------------------------------------------------------------------------
__global__ void ctx_kernel(const float* __restrict__ attnw,
                           const float* __restrict__ V,
                           float* __restrict__ ctx) {
  int t = blockIdx.x * blockDim.x + threadIdx.x;
  if (t >= BSZ * 512) return;
  int b = t >> 9, c = t & 511;
  int hh = c >> 7;
  const float* aw = attnw + (size_t)b * 16 + hh * 4;
  float o = 0.0f;
#pragma unroll
  for (int n = 0; n < 4; ++n)
    o = fmaf(aw[n], V[((size_t)b * 4 + n) * 512 + c], o);
  ctx[t] = o;
}

// ---------------------------------------------------------------------------
// Host side
// ---------------------------------------------------------------------------
extern "C" void kernel_launch(void* const* d_in, const int* in_sizes, int n_in,
                              void* d_out, int out_size, void* d_ws, size_t ws_size,
                              hipStream_t stream) {
  (void)in_sizes; (void)n_in; (void)out_size; (void)ws_size;

  const float* hidden = (const float*)d_in[10];
  const int*   nbi    = (const int*)  d_in[8];
  const float* ndis   = (const float*)d_in[9];

  float* out = (float*)d_out;
  float* q2_out = out;                         // [BS,8]
  float* h_out  = out + (size_t)BSZ * 8;       // [BS,256]
  float* ae_out = out + (size_t)BSZ * 264;     // [BS,512] all_emb
  float* p2_out = out + (size_t)BSZ * 776;     // [BS,8]

  // -------- workspace layout (floats), hand-overlapped across phases -------
  float* W = (float*)d_ws;
  float* ENC   = W + 0;                        // BS*224   (phase 1)
  float* X     = W + 7340032;                  // BS*256
  float* GI    = W + 15728640;                 // BS*768
  float* GH    = W + 40894464;                 // BS*768  (ends 66060288)
  float* NBCAT = W + 0;                        // BS*4*288 (reuses ENC/X/GI)
  float* CTXb  = W + 0;                        // BS*512   (reuses NBCAT)
  float* PQ    = W + 16777216;                 // BS*512   (p1|q1)
  float* NB    = W + 37748736;                 // BS*4*256 (reuses GI tail/GH)
  float* Qb    = W + 71303168;                 // BS*512
  float* Kb    = W + 88080384;                 // BS*4*512
  float* Vb    = W + 155189248;                // BS*4*512
  float* ATTN  = W + 222298112;                // BS*16
  unsigned short* WB = (unsigned short*)(W + 222822400); // bf16 weights

  const int O_FC1 = 0,       O_IH = 57344,   O_HH = 253952, O_NEI = 450560;
  const int O_EMB = 524288,  O_WQ = 589824,  O_WK = 720896, O_WV = 851968;
  const int O_WO  = 983040,  O_P1 = 1114112, O_Q1 = 1245184;
  const int O_Q2  = 1376256, O_P2 = 1384448;

  auto prep = [&](int srcIdx, int woff, int Keff, int Kpad, int Npad, int Neff,
                  int stride, int tr) {
    int total = (Kpad >> 5) * (Npad >> 4) * 512;
    prep_weight<<<(total + 255) / 256, 256, 0, stream>>>(
        (const float*)d_in[srcIdx], WB + woff, Keff, Kpad, Npad, Neff, stride, tr);
  };
  auto gemm = [&](const float* A, int woff, int biasIdx, float* C,
                  int M, int Kpad, int Npad, int Nout, int lda, int ldc, int act) {
    if ((Npad & 63) == 0) {
      long waves = (long)(M >> 4) * (Npad >> 6);
      int blocks = (int)((waves + 7) / 8);
      gemm_bf16_wmma<4><<<blocks, 256, 0, stream>>>(
          A, WB + woff, (const float*)d_in[biasIdx], C,
          M, Kpad, Npad, Nout, lda, ldc, act);
    } else {
      long waves = (long)(M >> 4) * (Npad >> 4);
      int blocks = (int)((waves + 7) / 8);
      gemm_bf16_wmma<1><<<blocks, 256, 0, stream>>>(
          A, WB + woff, (const float*)d_in[biasIdx], C,
          M, Kpad, Npad, Nout, lda, ldc, act);
    }
  };

  // -------- weight prep (tiny; re-done each call for determinism) ---------
  prep(25, O_FC1, 224, 224, 256, 256, 256, 0);   // w_fc1
  prep(27, O_IH,  256, 256, 768, 768, 256, 1);   // w_ih  (transposed use)
  prep(28, O_HH,  256, 256, 768, 768, 256, 1);   // w_hh
  prep(33, O_NEI, 264, 288, 256, 256, 256, 0);   // w_neigh (K padded 264->288)
  prep(35, O_EMB, 256, 256, 256, 256, 256, 0);   // w_emb
  prep(37, O_WQ,  256, 256, 512, 512, 512, 0);   // wq
  prep(39, O_WK,  256, 256, 512, 512, 512, 0);   // wk
  prep(41, O_WV,  256, 256, 512, 512, 512, 0);   // wv
  prep(43, O_WO,  512, 512, 256, 256, 256, 0);   // wo
  prep(45, O_P1,  512, 512, 256, 256, 256, 0);   // w_p1
  prep(49, O_Q1,  512, 512, 256, 256, 256, 0);   // w_q1
  prep(51, O_Q2,  512, 512,  16,   8,   8, 0);   // w_q2 (N padded 8->16)
  prep(47, O_P2,  256, 256,  16,   8,   8, 0);   // w_p2

  // -------- encoder ---------------------------------------------------------
  EncArgs ea;
  for (int i = 0; i < 6; ++i) {
    ea.x[i] = (const float*)d_in[i];
    ea.w[i] = (const float*)d_in[11 + 2 * i];
    ea.b[i] = (const float*)d_in[12 + 2 * i];
  }
  ea.phase_emb = (const float*)d_in[23];
  ea.mask_emb  = (const float*)d_in[24];
  ea.cur = (const int*)d_in[6];
  ea.msk = (const int*)d_in[7];
  ea.enc = ENC;
  encode_kernel<<<(BSZ * 224 + 255) / 256, 256, 0, stream>>>(ea);

  // x = relu(enc @ w_fc1 + b_fc1)
  gemm(ENC, O_FC1, 26, X, BSZ, 224, 256, 256, 224, 256, 1);
  // GRU gates
  gemm(X,      O_IH, 29, GI, BSZ, 256, 768, 768, 256, 768, 0);
  gemm(hidden, O_HH, 30, GH, BSZ, 256, 768, 768, 256, 768, 0);
  gru_kernel<<<(BSZ * 256 + 255) / 256, 256, 0, stream>>>(GI, GH, hidden, h_out);

  // neighbor gather + MLP
  gather_kernel<<<(BSZ * 4 * 288 + 255) / 256, 256, 0, stream>>>(
      h_out, nbi, ndis, (const float*)d_in[31], (const float*)d_in[32], NBCAT);
  gemm(NBCAT, O_NEI, 34, NB, BSZ * 4, 288, 256, 256, 288, 256, 1);

  // agent = relu(h @ w_emb + b) -> all_emb[:, :256]
  gemm(h_out, O_EMB, 36, ae_out, BSZ, 256, 256, 256, 256, 512, 1);

  // Q/K/V projections
  gemm(ae_out, O_WQ, 38, Qb, BSZ,     256, 512, 512, 512, 512, 0);
  gemm(NB,     O_WK, 40, Kb, BSZ * 4, 256, 512, 512, 256, 512, 0);
  gemm(NB,     O_WV, 42, Vb, BSZ * 4, 256, 512, 512, 256, 512, 0);

  // masked softmax attention over 4 neighbors
  attn_kernel<<<(BSZ * 4 + 255) / 256, 256, 0, stream>>>(Qb, Kb, nbi, ATTN);
  ctx_kernel<<<(BSZ * 512 + 255) / 256, 256, 0, stream>>>(ATTN, Vb, CTXb);

  // neigh_out = relu(ctx @ wo + bo) -> all_emb[:, 256:]
  gemm(CTXb, O_WO, 44, ae_out + 256, BSZ, 512, 256, 256, 512, 512, 1);

  // heads: p1 | q1 packed into PQ [BS,512]
  gemm(ae_out, O_P1, 46, PQ,       BSZ, 512, 256, 256, 512, 512, 1);
  gemm(ae_out, O_Q1, 50, PQ + 256, BSZ, 512, 256, 256, 512, 512, 1);
  // q2 = [p1|q1] @ w_q2 + b ; p2 = p1 @ w_p2 + b
  gemm(PQ, O_Q2, 52, q2_out, BSZ, 512, 16, 8, 512, 8, 0);
  gemm(PQ, O_P2, 48, p2_out, BSZ, 256, 16, 8, 512, 8, 0);
}
